// Qwen3MoeForCausalLM_53042846105765
// MI455X (gfx1250) — matrix operand order repaired and checked by reading, not compiled, not run on hardware
//
#include <hip/hip_runtime.h>
#include <math.h>

#define TT   2048   // tokens
#define HH   2048   // hidden
#define II   768    // moe intermediate
#define EE   32     // experts
#define TOPK 8
#define CAPE 1024   // per-expert capacity

typedef __attribute__((ext_vector_type(16))) __bf16 v16bf;
typedef __attribute__((ext_vector_type(8)))  __bf16 v8bf;
typedef __attribute__((ext_vector_type(8)))  float  v8f;

// ---------------- workspace layout (bytes) ----------------
static constexpr size_t OFF_WGUT = 0;                                    // bf16 [E][2I][H]
static constexpr size_t SZ_WGUT  = (size_t)EE * 2 * II * HH * 2;
static constexpr size_t OFF_WDT  = OFF_WGUT + SZ_WGUT;                   // bf16 [E][H][I]
static constexpr size_t SZ_WDT   = (size_t)EE * HH * II * 2;
static constexpr size_t OFF_XBF  = OFF_WDT + SZ_WDT;                     // bf16 [T][H]
static constexpr size_t SZ_XBF   = (size_t)TT * HH * 2;
static constexpr size_t OFF_H    = OFF_XBF + SZ_XBF;                     // bf16 [E][CAP][I]
static constexpr size_t SZ_H     = (size_t)EE * CAPE * II * 2;
static constexpr size_t OFF_TOK  = OFF_H + SZ_H;                         // int  [E][CAP]
static constexpr size_t SZ_TOK   = (size_t)EE * CAPE * 4;
static constexpr size_t OFF_WGT  = OFF_TOK + SZ_TOK;                     // f32  [E][CAP]
static constexpr size_t SZ_WGT   = (size_t)EE * CAPE * 4;
static constexpr size_t OFF_CNT  = OFF_WGT + SZ_WGT;                     // int  [E]
static constexpr size_t SZ_CNT   = (size_t)EE * 4;
static constexpr size_t SZ_ROUTE = SZ_TOK + SZ_WGT + SZ_CNT;             // zeroed together

// ---------------- utility kernels ----------------
__global__ void zero16_kernel(uint4* __restrict__ p, size_t n16) {
  size_t i = (size_t)blockIdx.x * blockDim.x + threadIdx.x;
  uint4 z; z.x = 0; z.y = 0; z.z = 0; z.w = 0;
  for (; i < n16; i += (size_t)gridDim.x * blockDim.x) p[i] = z;
}

__global__ void cvt_f32_bf16_kernel(const float* __restrict__ in, __bf16* __restrict__ out, size_t n) {
  size_t i = (size_t)blockIdx.x * blockDim.x + threadIdx.x;
  for (; i < n; i += (size_t)gridDim.x * blockDim.x) out[i] = (__bf16)in[i];
}

// Transpose + convert: fp32 in[R][C] (batched over z) -> bf16 out[C][R]
__global__ __launch_bounds__(256) void transpose_cvt_kernel(const float* __restrict__ in,
                                                            __bf16* __restrict__ out,
                                                            int R, int C) {
  const int e = blockIdx.z;
  in  += (size_t)e * R * C;
  out += (size_t)e * R * C;
  __shared__ float tile[32][33];
  const int c0 = blockIdx.x * 32, r0 = blockIdx.y * 32;
  const int tx = threadIdx.x, ty = threadIdx.y;   // 32 x 8
#pragma unroll
  for (int i = 0; i < 4; ++i)
    tile[ty + i * 8][tx] = in[(size_t)(r0 + ty + i * 8) * C + (c0 + tx)];
  __syncthreads();
#pragma unroll
  for (int i = 0; i < 4; ++i)
    out[(size_t)(c0 + ty + i * 8) * R + (r0 + tx)] = (__bf16)tile[tx][ty + i * 8];
}

// ---------------- router: logits -> softmax -> top-8 -> renorm -> permute ----------------
// One wave (32 lanes == 32 experts) handles 8 tokens.
__global__ __launch_bounds__(32) void router_kernel(const float* __restrict__ x,
                                                    const float* __restrict__ gw,
                                                    const float* __restrict__ gb,
                                                    int* __restrict__ tok,
                                                    float* __restrict__ wgt,
                                                    int* __restrict__ counts) {
  const int lane = threadIdx.x;        // expert id
  const int t0 = blockIdx.x * 8;
  float acc[8];
  const float b = gb[lane];
#pragma unroll
  for (int j = 0; j < 8; ++j) acc[j] = b;
  for (int k = 0; k < HH; ++k) {
    const float w = gw[k * EE + lane];           // coalesced across lanes
#pragma unroll
    for (int j = 0; j < 8; ++j)                  // x reads are wave-uniform (scalar)
      acc[j] = fmaf(x[(size_t)(t0 + j) * HH + k], w, acc[j]);
  }
  for (int j = 0; j < 8; ++j) {
    const int t = t0 + j;
    float l = acc[j];
    float m = l;
    for (int off = 16; off; off >>= 1) m = fmaxf(m, __shfl_xor(m, off, 32));
    float p = __expf(l - m);
    float s = p;
    for (int off = 16; off; off >>= 1) s += __shfl_xor(s, off, 32);
    p /= s;
    const float orig = p;
    float topsum = 0.f;
    int sel = 0;
    for (int i = 0; i < TOPK; ++i) {
      float v = p; int who = lane;
      for (int off = 16; off; off >>= 1) {
        float ov = __shfl_xor(v, off, 32);
        int   ow = __shfl_xor(who, off, 32);
        if (ov > v || (ov == v && ow < who)) { v = ov; who = ow; }
      }
      topsum += v;
      if (lane == who) { sel = 1; p = -1.f; }
    }
    if (sel) {
      int pos = atomicAdd(&counts[lane], 1);
      if (pos < CAPE) {
        tok[lane * CAPE + pos] = t;
        wgt[lane * CAPE + pos] = orig / topsum;
      }
    }
  }
}

// ---------------- GEMM1: h = silu(Xe @ Wg) * (Xe @ Wu), bf16 WMMA ----------------
// Register-blocked: each wave computes a 64-row x (32 gate + 32 up) output block.
// grid: (CAP/64, 24/8, E), block 256 (8 waves).
__global__ __launch_bounds__(256) void moe_gemm1_kernel(const __bf16* __restrict__ xbf,
                                                        const __bf16* __restrict__ wgu_t,
                                                        const int* __restrict__ tok,
                                                        const int* __restrict__ counts,
                                                        __bf16* __restrict__ h) {
  const int e = blockIdx.z;
  const int m0 = blockIdx.x * 64;
  if (m0 >= counts[e]) return;                         // uniform across block
  const int wave = threadIdx.x >> 5;
  const int lane = threadIdx.x & 31;
  const int hi = lane >> 4, ln = lane & 15;
  const int n0 = (blockIdx.y * 8 + wave) * 32;         // gate column block (32 wide)

  const __bf16* Ap[4];
#pragma unroll
  for (int mi = 0; mi < 4; ++mi) {
    const int row = tok[e * CAPE + m0 + mi * 16 + ln]; // A: lane holds m = lane%16
    Ap[mi] = xbf + (size_t)row * HH;
  }
  const __bf16* Bg0 = wgu_t + ((size_t)e * (2 * II) + (n0 + ln)) * HH;
  const __bf16* Bu0 = Bg0 + (size_t)II * HH;

  v8f ag[4][2], au[4][2];
#pragma unroll
  for (int mi = 0; mi < 4; ++mi)
#pragma unroll
    for (int ni = 0; ni < 2; ++ni) { ag[mi][ni] = (v8f){}; au[mi][ni] = (v8f){}; }

  for (int k0 = 0; k0 < HH; k0 += 32) {
    v16bf a[4];
#pragma unroll
    for (int mi = 0; mi < 4; ++mi) {
      v8bf lo = *(const v8bf*)(Ap[mi] + k0 + hi * 16);
      v8bf hi8 = *(const v8bf*)(Ap[mi] + k0 + hi * 16 + 8);
      a[mi] = __builtin_shufflevector(lo, hi8, 0,1,2,3,4,5,6,7,8,9,10,11,12,13,14,15);
    }
    v16bf bg[2], bu[2];
#pragma unroll
    for (int ni = 0; ni < 2; ++ni) {
      bg[ni] = *(const v16bf*)(Bg0 + (size_t)ni * 16 * HH + k0 + hi * 16);
      bu[ni] = *(const v16bf*)(Bu0 + (size_t)ni * 16 * HH + k0 + hi * 16);
    }
#pragma unroll
    for (int mi = 0; mi < 4; ++mi)
#pragma unroll
      for (int ni = 0; ni < 2; ++ni) {
        ag[mi][ni] = __builtin_amdgcn_wmma_f32_16x16x32_bf16(false, a[mi], false, bg[ni], (short)0, ag[mi][ni], false, false);
        au[mi][ni] = __builtin_amdgcn_wmma_f32_16x16x32_bf16(false, a[mi], false, bu[ni], (short)0, au[mi][ni], false, false);
      }
  }
  __bf16* hp = h + (size_t)e * CAPE * II;
#pragma unroll
  for (int mi = 0; mi < 4; ++mi)
#pragma unroll
    for (int ni = 0; ni < 2; ++ni)
#pragma unroll
      for (int v = 0; v < 8; ++v) {
        const int m = m0 + mi * 16 + v + 8 * hi;       // C layout: M = v + 8*(lane/16)
        const float g = ag[mi][ni][v];
        const float val = (g / (1.f + __expf(-g))) * au[mi][ni][v];
        hp[(size_t)m * II + (n0 + ni * 16 + ln)] = (__bf16)val;
      }
}

// ---------------- GEMM2: out += w * (h @ Wd), bf16 WMMA + f32 atomics ----------------
// Register-blocked: each wave computes a 64x64 output block.
// grid: (CAP/64, (H/64)/8, E), block 256.
__global__ __launch_bounds__(256) void moe_gemm2_kernel(const __bf16* __restrict__ h,
                                                        const __bf16* __restrict__ wd_t,
                                                        const int* __restrict__ tok,
                                                        const float* __restrict__ wgt,
                                                        const int* __restrict__ counts,
                                                        float* __restrict__ out) {
  const int e = blockIdx.z;
  const int m0 = blockIdx.x * 64;
  if (m0 >= counts[e]) return;
  const int wave = threadIdx.x >> 5;
  const int lane = threadIdx.x & 31;
  const int hi = lane >> 4, ln = lane & 15;
  const int n0 = (blockIdx.y * 8 + wave) * 64;

  const __bf16* A0 = h + (size_t)(e * CAPE + m0 + ln) * II;   // +mi*16 rows
  const __bf16* B0 = wd_t + ((size_t)e * HH + (n0 + ln)) * II; // +ni*16 rows

  v8f acc[4][4];
#pragma unroll
  for (int mi = 0; mi < 4; ++mi)
#pragma unroll
    for (int ni = 0; ni < 4; ++ni) acc[mi][ni] = (v8f){};

  for (int k0 = 0; k0 < II; k0 += 32) {
    v16bf a[4];
#pragma unroll
    for (int mi = 0; mi < 4; ++mi) {
      const __bf16* Ap = A0 + (size_t)mi * 16 * II;
      v8bf lo = *(const v8bf*)(Ap + k0 + hi * 16);
      v8bf hi8 = *(const v8bf*)(Ap + k0 + hi * 16 + 8);
      a[mi] = __builtin_shufflevector(lo, hi8, 0,1,2,3,4,5,6,7,8,9,10,11,12,13,14,15);
    }
    v16bf bm[4];
#pragma unroll
    for (int ni = 0; ni < 4; ++ni)
      bm[ni] = *(const v16bf*)(B0 + (size_t)ni * 16 * II + k0 + hi * 16);
#pragma unroll
    for (int mi = 0; mi < 4; ++mi)
#pragma unroll
      for (int ni = 0; ni < 4; ++ni)
        acc[mi][ni] = __builtin_amdgcn_wmma_f32_16x16x32_bf16(false, a[mi], false, bm[ni], (short)0, acc[mi][ni], false, false);
  }
#pragma unroll
  for (int mi = 0; mi < 4; ++mi)
#pragma unroll
    for (int v = 0; v < 8; ++v) {
      const int m = m0 + mi * 16 + v + 8 * hi;
      const float wv = wgt[e * CAPE + m];              // 0 on padded rows
      const int t = tok[e * CAPE + m];
      float* op = out + (size_t)t * HH + n0 + ln;
#pragma unroll
      for (int ni = 0; ni < 4; ++ni)
        unsafeAtomicAdd(op + ni * 16, acc[mi][ni][v] * wv);
    }
}

// ---------------- launch ----------------
extern "C" void kernel_launch(void* const* d_in, const int* in_sizes, int n_in,
                              void* d_out, int out_size, void* d_ws, size_t ws_size,
                              hipStream_t stream) {
  (void)in_sizes; (void)n_in; (void)out_size; (void)ws_size;
  const float* x   = (const float*)d_in[0];   // [T,H]
  const float* gw  = (const float*)d_in[1];   // [H,E]
  const float* gb  = (const float*)d_in[2];   // [E]
  const float* wgu = (const float*)d_in[3];   // [E,H,2I]
  const float* wd  = (const float*)d_in[4];   // [E,I,H]
  float* out = (float*)d_out;                 // [T,H]

  char* ws = (char*)d_ws;
  __bf16* wgu_t = (__bf16*)(ws + OFF_WGUT);
  __bf16* wd_t  = (__bf16*)(ws + OFF_WDT);
  __bf16* xbf   = (__bf16*)(ws + OFF_XBF);
  __bf16* hbuf  = (__bf16*)(ws + OFF_H);
  int*    tok   = (int*)   (ws + OFF_TOK);
  float*  wgt   = (float*) (ws + OFF_WGT);
  int*    cnt   = (int*)   (ws + OFF_CNT);

  // 1) zero output and routing buffers (tok/wgt/counts are contiguous)
  zero16_kernel<<<2048, 256, 0, stream>>>((uint4*)out, (size_t)TT * HH * 4 / 16);
  zero16_kernel<<<64, 256, 0, stream>>>((uint4*)(ws + OFF_TOK), SZ_ROUTE / 16);

  // 2) x -> bf16
  cvt_f32_bf16_kernel<<<4096, 256, 0, stream>>>(x, xbf, (size_t)TT * HH);

  // 3) weight convert+transpose: wgu [E][H][2I] -> [E][2I][H]; wd [E][I][H] -> [E][H][I]
  transpose_cvt_kernel<<<dim3(2 * II / 32, HH / 32, EE), dim3(32, 8), 0, stream>>>(wgu, wgu_t, HH, 2 * II);
  transpose_cvt_kernel<<<dim3(HH / 32, II / 32, EE), dim3(32, 8), 0, stream>>>(wd, wd_t, II, HH);

  // 4) router
  router_kernel<<<TT / 8, 32, 0, stream>>>(x, gw, gb, tok, wgt, cnt);

  // 5) grouped GEMM1 (gate/up fused with SiLU*mul), 64x64 per wave
  moe_gemm1_kernel<<<dim3(CAPE / 64, (II / 32) / 8, EE), 256, 0, stream>>>(xbf, wgu_t, tok, cnt, hbuf);

  // 6) grouped GEMM2 + weighted scatter-add combine, 64x64 per wave
  moe_gemm2_kernel<<<dim3(CAPE / 64, (HH / 64) / 8, EE), 256, 0, stream>>>(hbuf, wd_t, tok, wgt, cnt, out);
}
